// DeltaNet2_12472585027975
// MI455X (gfx1250) — compile-verified, matrix operand order checked by reading, not run on hardware
//
#include <hip/hip_runtime.h>
#include <hip/hip_bf16.h>

// ---------------------------------------------------------------------------
// DeltaNet block for MI455X (gfx1250, wave32, WMMA).
// bf16 WMMA (v_wmma_f32_16x16x32_bf16) with fp32 accumulation.
// All LDS layouts are chosen so every WMMA operand fragment is two contiguous
// ds_load_b128's (A operands row-major [M][K]; B operands transposed [N][K]).
// ---------------------------------------------------------------------------

#define B_   2
#define T_   4096
#define D_   2048
#define H_   16
#define DK   128
#define CH   64

typedef __bf16 bf16_t;
typedef __bf16 v16bf __attribute__((ext_vector_type(16)));
typedef __bf16 v8bf  __attribute__((ext_vector_type(8)));
typedef __bf16 v4bf  __attribute__((ext_vector_type(4)));
typedef float  v8f   __attribute__((ext_vector_type(8)));

// Fragment layout per ISA 7.12.2 (16x32 bf16 operand): lane half 0 carries
// K = {0..7, 16..23}, half 1 carries K = {8..15, 24..31}. Both 8-element runs
// are contiguous -> two 16B LDS loads.
__device__ __forceinline__ v16bf ld2x8(const bf16_t* base) {
  v8bf lo = *reinterpret_cast<const v8bf*>(base);
  v8bf hi = *reinterpret_cast<const v8bf*>(base + 16);
  return __builtin_shufflevector(lo, hi, 0, 1, 2, 3, 4, 5, 6, 7,
                                 8, 9, 10, 11, 12, 13, 14, 15);
}

// A operand, source row-major [M][K], leading dim ld (K contiguous).
__device__ __forceinline__ v16bf ldA(const bf16_t* p, int ld, int m0, int k0, int lane) {
  const int half = (lane >> 4) & 1, r = lane & 15;
  return ld2x8(p + (size_t)(m0 + r) * ld + k0 + half * 8);
}

// B operand = X^T where X stored row-major [N][K], leading dim ld (K contiguous).
__device__ __forceinline__ v16bf ldBt(const bf16_t* p, int ld, int k0, int n0, int lane) {
  const int half = (lane >> 4) & 1, r = lane & 15;
  return ld2x8(p + (size_t)(n0 + r) * ld + k0 + half * 8);
}

__device__ __forceinline__ v8f wmma_bf16(v16bf a, v16bf b, v8f c) {
  return __builtin_amdgcn_wmma_f32_16x16x32_bf16(false, a, false, b, (short)0, c,
                                                 false, false);
}

// ---------------------------------------------------------------------------
// GEMM: C[M,N] = A[M,K] @ B[K,N], fp32 in/out, bf16 WMMA inside.
// Block tile 128x128, 8 waves of 32x64, K-tile 64 (2 WMMA K-steps per stage).
// B is staged TRANSPOSED into LDS so its fragments load contiguously.
// ---------------------------------------------------------------------------
__global__ __launch_bounds__(256) void gemm_bf16(const float* __restrict__ A,
                                                 const float* __restrict__ B,
                                                 float* __restrict__ C,
                                                 int M, int N, int K) {
  __shared__ bf16_t As[128][64];   // [m][k]
  __shared__ bf16_t Bs[128][64];   // [n][k]  (transposed tile of B)
  const int tid = threadIdx.x, lane = tid & 31, wv = tid >> 5;
  const int bm = blockIdx.y * 128, bn = blockIdx.x * 128;
  const int wm = (wv >> 1) * 32, wn = (wv & 1) * 64;
  const v8f vzero = {};
  v8f acc[2][4];
#pragma unroll
  for (int i = 0; i < 2; ++i)
#pragma unroll
    for (int j = 0; j < 4; ++j) acc[i][j] = vzero;

  for (int k0 = 0; k0 < K; k0 += 64) {
    if (k0 + 64 < K) {  // speculative prefetch of next tiles (global_prefetch_b8)
      __builtin_prefetch(&A[(size_t)(bm + (tid & 127)) * K + k0 + 64], 0, 0);
      __builtin_prefetch(&B[(size_t)(k0 + 64 + (tid & 63)) * N + bn], 0, 0);
    }
    // A tile: 128x64 f32 -> bf16, row-major (coalesced float4 reads).
#pragma unroll
    for (int it = 0; it < 8; ++it) {
      int idx = tid + it * 256;             // 2048 float4 quads
      int r = idx >> 4, c4 = (idx & 15) << 2;
      const float4 va =
          *reinterpret_cast<const float4*>(&A[(size_t)(bm + r) * K + k0 + c4]);
      v4bf pk;
      pk[0] = (bf16_t)va.x; pk[1] = (bf16_t)va.y;
      pk[2] = (bf16_t)va.z; pk[3] = (bf16_t)va.w;
      *reinterpret_cast<v4bf*>(&As[r][c4]) = pk;
    }
    // B tile transposed: thread owns (n, k-quad); global reads stay coalesced
    // along n for each k; LDS store is one packed b64.
#pragma unroll
    for (int it = 0; it < 8; ++it) {
      int idx = tid + it * 256;             // 2048 quads
      int n = idx & 127, kq = (idx >> 7) << 2;   // kq in {0,4,...,60}
      v4bf pk;
#pragma unroll
      for (int i = 0; i < 4; ++i)
        pk[i] = (bf16_t)B[(size_t)(k0 + kq + i) * N + bn + n];
      *reinterpret_cast<v4bf*>(&Bs[n][kq]) = pk;
    }
    __syncthreads();
#pragma unroll
    for (int kin = 0; kin < 64; kin += 32) {
      v16bf af[2], bfr[4];
#pragma unroll
      for (int i = 0; i < 2; ++i) af[i] = ldA(&As[0][0], 64, wm + 16 * i, kin, lane);
#pragma unroll
      for (int j = 0; j < 4; ++j) bfr[j] = ldBt(&Bs[0][0], 64, kin, wn + 16 * j, lane);
#pragma unroll
      for (int i = 0; i < 2; ++i)
#pragma unroll
        for (int j = 0; j < 4; ++j) acc[i][j] = wmma_bf16(af[i], bfr[j], acc[i][j]);
    }
    __syncthreads();
  }
  const int half = (lane >> 4) & 1, r = lane & 15;
#pragma unroll
  for (int i = 0; i < 2; ++i)
#pragma unroll
    for (int j = 0; j < 4; ++j)
#pragma unroll
      for (int e = 0; e < 8; ++e)
        C[(size_t)(bm + wm + 16 * i + e + half * 8) * N + bn + wn + 16 * j + r] =
            acc[i][j][e];
}

// ---------------------------------------------------------------------------
// beta = sigmoid(x @ Wb), [B*T, 16]. One 64-thread block per row.
// ---------------------------------------------------------------------------
__global__ __launch_bounds__(64) void beta_kernel(const float* __restrict__ x,
                                                  const float* __restrict__ Wb,
                                                  float* __restrict__ beta) {
  const int row = blockIdx.x;
  const int tid = threadIdx.x;
  const int h = tid & 15, part = tid >> 4;
  const float* xr = x + (size_t)row * D_;
  float acc = 0.f;
  for (int d = part; d < D_; d += 4) acc += xr[d] * Wb[(size_t)d * H_ + h];
  __shared__ float red[64];
  red[tid] = acc;
  __syncthreads();
  if (part == 0) {
    float s = red[h] + red[h + 16] + red[h + 32] + red[h + 48];
    beta[(size_t)row * H_ + h] = 1.f / (1.f + __expf(-s));
  }
}

// ---------------------------------------------------------------------------
// Causal shared 4-tap conv + SiLU, elementwise over [B,T,D].
// ---------------------------------------------------------------------------
__global__ __launch_bounds__(256) void conv_silu_kernel(const float* __restrict__ in,
                                                        const float* __restrict__ w,
                                                        float* __restrict__ out) {
  size_t idx = (size_t)blockIdx.x * 256 + threadIdx.x;
  int c = (int)(idx & (D_ - 1));
  size_t bt = idx >> 11;  // / D_
  int t = (int)(bt & (T_ - 1));
  float acc = 0.f;
#pragma unroll
  for (int j = 0; j < 4; ++j) {
    int tj = t + j - 3;
    if (tj >= 0) acc += w[j] * in[(bt + (size_t)(j - 3)) * D_ + c];
  }
  float sg = 1.f / (1.f + __expf(-acc));
  out[idx] = acc * sg;
}

// ---------------------------------------------------------------------------
// Per-(b,t,h) L2 norm over 128 dims (wave32 per vector), optional extra scale.
// ---------------------------------------------------------------------------
__global__ __launch_bounds__(256) void l2norm_kernel(float* __restrict__ v,
                                                     float scale, int nvec) {
  int gw = (int)(((size_t)blockIdx.x * 256 + threadIdx.x) >> 5);
  int lane = threadIdx.x & 31;
  if (gw >= nvec) return;
  float* p = v + (size_t)gw * DK;
  float4 xv = *reinterpret_cast<float4*>(p + lane * 4);
  float ss = xv.x * xv.x + xv.y * xv.y + xv.z * xv.z + xv.w * xv.w;
#pragma unroll
  for (int m = 16; m >= 1; m >>= 1) ss += __shfl_xor(ss, m, 32);
  float rr = rsqrtf(ss + 1e-6f) * scale;
  xv.x *= rr; xv.y *= rr; xv.z *= rr; xv.w *= rr;
  *reinterpret_cast<float4*>(p + lane * 4) = xv;
}

// ---------------------------------------------------------------------------
// Per-(b,t,h) RMSNorm over 128 dims with weight.
// ---------------------------------------------------------------------------
__global__ __launch_bounds__(256) void rms_kernel(float* __restrict__ v,
                                                  const float* __restrict__ w,
                                                  int nvec) {
  int gw = (int)(((size_t)blockIdx.x * 256 + threadIdx.x) >> 5);
  int lane = threadIdx.x & 31;
  if (gw >= nvec) return;
  float* p = v + (size_t)gw * DK;
  float4 xv = *reinterpret_cast<float4*>(p + lane * 4);
  float ss = xv.x * xv.x + xv.y * xv.y + xv.z * xv.z + xv.w * xv.w;
#pragma unroll
  for (int m = 16; m >= 1; m >>= 1) ss += __shfl_xor(ss, m, 32);
  float rr = rsqrtf(ss * (1.f / DK) + 1e-5f);
  const float4 wv = *reinterpret_cast<const float4*>(w + lane * 4);
  xv.x *= rr * wv.x; xv.y *= rr * wv.y; xv.z *= rr * wv.z; xv.w *= rr * wv.w;
  *reinterpret_cast<float4*>(p + lane * 4) = xv;
}

// ---------------------------------------------------------------------------
// Chunked delta rule. One block per (b,h); sequential over 64 chunks.
// State kept TRANSPOSED: SfT/SbT = S^T [dv][dk]; u kept as ubT = u^T [dv][tok];
// extra kbT = k^T [dk][tok] copy so every WMMA operand loads contiguously.
// Dynamic LDS ~216 KB (320 KB/WGP on CDNA5).
// ---------------------------------------------------------------------------
__global__ __launch_bounds__(256) void delta_kernel(const float* __restrict__ qg,
                                                    const float* __restrict__ kg,
                                                    const float* __restrict__ vg,
                                                    const float* __restrict__ betag,
                                                    float* __restrict__ og) {
  extern __shared__ __align__(16) char smem_raw[];
  float*  SfT   = (float*)smem_raw;                 // [dv][dk] f32   65536 B
  bf16_t* SbT   = (bf16_t*)(SfT + DK * DK);         // [dv][dk] bf16  32768 B
  bf16_t* qb    = SbT + DK * DK;                    // [tok][dk]      16384 B
  bf16_t* kb    = qb + CH * DK;                     // [tok][dk]      16384 B
  bf16_t* kbb   = kb + CH * DK;                     // [tok][dk] k*b  16384 B
  bf16_t* kbT   = kbb + CH * DK;                    // [dk][tok]      16384 B
  bf16_t* wbuf  = kbT + DK * CH;                    // [tok][dk]      16384 B
  bf16_t* ubT   = wbuf + CH * DK;                   // [dv][tok]      16384 B
  bf16_t* attnb = ubT + DK * CH;                    // [tok][tok]      8192 B
  float*  Mf    = (float*)(attnb + CH * CH);        // [tok][tok] f32 16384 B
  float*  betaS = Mf + CH * CH;                     // [tok]            256 B

  const int tid = threadIdx.x, lane = tid & 31, wv = tid >> 5;
  const int half = (lane >> 4) & 1, r = lane & 15;
  const int bh = blockIdx.x, bb = bh / H_, h = bh % H_;
  const size_t headoff = (size_t)h * DK;
  const v8f vzero = {};

  for (int i = tid; i < DK * DK; i += 256) {
    SfT[i] = 0.f;
    SbT[i] = (bf16_t)0.f;
  }
  __syncthreads();

  for (int ci = 0; ci < T_ / CH; ++ci) {
    const size_t rowbase = (size_t)bb * T_ + ci * CH;

    // ---- stage beta ----
    if (tid < CH) betaS[tid] = betag[(rowbase + tid) * H_ + h];
    __syncthreads();

    // ---- stage q, k, k*beta (row-major) ----
    for (int idx = tid; idx < CH * DK; idx += 256) {
      int rr2 = idx >> 7, c = idx & (DK - 1);
      size_t g = (rowbase + rr2) * D_ + headoff + c;
      qb[idx] = (bf16_t)qg[g];
      float kv = kg[g];
      kb[idx] = (bf16_t)kv;
      kbb[idx] = (bf16_t)(kv * betaS[rr2]);
    }
    // ---- stage kbT = k^T [dk][tok] with packed b64 stores ----
    for (int idx = tid; idx < DK * (CH / 4); idx += 256) {  // 2048 quads
      int c = idx & (DK - 1), tq = (idx >> 7) << 2;         // tq in {0,4,..,60}
      v4bf pk;
#pragma unroll
      for (int i = 0; i < 4; ++i)
        pk[i] = (bf16_t)kg[(rowbase + tq + i) * D_ + headoff + c];
      *reinterpret_cast<v4bf*>(&kbT[c * CH + tq]) = pk;
    }
    __syncthreads();

    // ---- phase 1: Mf = (k*beta) @ k^T ; attn = tril(q @ k^T) ----
#pragma unroll
    for (int j = 0; j < 4; ++j) {
      int id = wv * 4 + j;
      int tm = (id & 15) >> 2, tn = id & 3;
      bool isAttn = id >= 16;
      v8f acc = vzero;
#pragma unroll
      for (int ks = 0; ks < 4; ++ks) {
        v16bf a = isAttn ? ldA(qb, DK, tm * 16, ks * 32, lane)
                         : ldA(kbb, DK, tm * 16, ks * 32, lane);
        v16bf b = ldBt(kb, DK, ks * 32, tn * 16, lane);
        acc = wmma_bf16(a, b, acc);
      }
#pragma unroll
      for (int e = 0; e < 8; ++e) {
        int grow = tm * 16 + e + half * 8, gcol = tn * 16 + r;
        if (isAttn)
          attnb[grow * CH + gcol] = (bf16_t)((gcol <= grow) ? acc[e] : 0.f);
        else
          Mf[grow * CH + gcol] = acc[e];
      }
    }
    __syncthreads();

    // ---- phase 2: forward substitution, column kept in registers ----
    // threads 0..127: u columns (rhs v*beta) -> ubT (contiguous stores)
    // threads 128..255: w columns (rhs k*beta) -> wbuf (strided stores)
    {
      const int c = tid & (DK - 1);
      const bool isU = tid < 128;
      const float* src = (isU ? vg : kg) + rowbase * D_ + headoff + c;
      bf16_t* dst = isU ? (ubT + (size_t)c * CH) : (wbuf + c);
      const int dstride = isU ? 1 : DK;
      float colv[CH];
#pragma unroll
      for (int i = 0; i < CH; ++i) {
        float acc = src[(size_t)i * D_] * betaS[i];
#pragma unroll
        for (int m = 0; m + 4 <= i; m += 4) {
          const float4 mv = *reinterpret_cast<const float4*>(&Mf[i * CH + m]);
          acc -= mv.x * colv[m] + mv.y * colv[m + 1] +
                 mv.z * colv[m + 2] + mv.w * colv[m + 3];
        }
#pragma unroll
        for (int m = i & ~3; m < i; ++m) acc -= Mf[i * CH + m] * colv[m];
        colv[i] = acc;
        dst[i * dstride] = (bf16_t)acc;
      }
    }
    __syncthreads();

    // ---- phase 3: u -= w @ S   (D tile read-modify-write in ubT, b128) ----
#pragma unroll
    for (int j = 0; j < 4; ++j) {
      int id = wv * 4 + j, tm = id >> 3, tn = id & 7;
      v8f acc = vzero;
#pragma unroll
      for (int ks = 0; ks < 4; ++ks) {
        v16bf a = ldA(wbuf, DK, tm * 16, ks * 32, lane);
        v16bf b = ldBt(SbT, DK, ks * 32, tn * 16, lane);
        acc = wmma_bf16(a, b, acc);
      }
      bf16_t* up = &ubT[(size_t)(tn * 16 + r) * CH + tm * 16 + half * 8];
      v8bf oldv = *reinterpret_cast<v8bf*>(up);
      v8bf newv;
#pragma unroll
      for (int e = 0; e < 8; ++e) newv[e] = (bf16_t)((float)oldv[e] - acc[e]);
      *reinterpret_cast<v8bf*>(up) = newv;
    }
    __syncthreads();

    // ---- phase 4: O = q @ S + attn @ u -> global ----
#pragma unroll
    for (int j = 0; j < 4; ++j) {
      int id = wv * 4 + j, tm = id >> 3, tn = id & 7;
      v8f acc = vzero;
#pragma unroll
      for (int ks = 0; ks < 4; ++ks) {
        v16bf a = ldA(qb, DK, tm * 16, ks * 32, lane);
        v16bf b = ldBt(SbT, DK, ks * 32, tn * 16, lane);
        acc = wmma_bf16(a, b, acc);
      }
#pragma unroll
      for (int ks = 0; ks < 2; ++ks) {
        v16bf a = ldA(attnb, CH, tm * 16, ks * 32, lane);
        v16bf b = ldBt(ubT, CH, ks * 32, tn * 16, lane);
        acc = wmma_bf16(a, b, acc);
      }
#pragma unroll
      for (int e = 0; e < 8; ++e) {
        int grow = tm * 16 + e + half * 8, gcol = tn * 16 + r;
        og[(rowbase + grow) * D_ + headoff + gcol] = acc[e];
      }
    }
    __syncthreads();

    // ---- phase 5: S^T += (k^T @ u)^T  i.e. SfT[dv][dk] tiles, vector C I/O --
#pragma unroll
    for (int j = 0; j < 8; ++j) {
      int id = wv * 8 + j, tm = id >> 3, tn = id & 7;  // tm: dk tile, tn: dv tile
      float* cp = &SfT[(size_t)(tn * 16 + r) * DK + tm * 16 + half * 8];
      const float4 ca = *reinterpret_cast<const float4*>(cp);
      const float4 cb = *reinterpret_cast<const float4*>(cp + 4);
      v8f acc;
      acc[0] = ca.x; acc[1] = ca.y; acc[2] = ca.z; acc[3] = ca.w;
      acc[4] = cb.x; acc[5] = cb.y; acc[6] = cb.z; acc[7] = cb.w;
#pragma unroll
      for (int ks = 0; ks < 2; ++ks) {
        v16bf a = ldA(kbT, CH, tm * 16, ks * 32, lane);   // A = k^T [dk][tok]
        v16bf b = ldBt(ubT, CH, ks * 32, tn * 16, lane);  // B = u   [tok][dv]
        acc = wmma_bf16(a, b, acc);
      }
      float4 oa, ob;
      oa.x = acc[0]; oa.y = acc[1]; oa.z = acc[2]; oa.w = acc[3];
      ob.x = acc[4]; ob.y = acc[5]; ob.z = acc[6]; ob.w = acc[7];
      *reinterpret_cast<float4*>(cp) = oa;
      *reinterpret_cast<float4*>(cp + 4) = ob;
      v8bf sb;
#pragma unroll
      for (int e = 0; e < 8; ++e) sb[e] = (bf16_t)acc[e];
      *reinterpret_cast<v8bf*>(&SbT[(size_t)(tn * 16 + r) * DK + tm * 16 + half * 8]) = sb;
    }
    __syncthreads();
  }
}

// ---------------------------------------------------------------------------
// Launcher
// ---------------------------------------------------------------------------
extern "C" void kernel_launch(void* const* d_in, const int* in_sizes, int n_in,
                              void* d_out, int out_size, void* d_ws, size_t ws_size,
                              hipStream_t stream) {
  (void)in_sizes; (void)n_in; (void)out_size; (void)ws_size;
  const float* x   = (const float*)d_in[0];
  const float* Wq  = (const float*)d_in[1];
  const float* Wk  = (const float*)d_in[2];
  const float* Wv  = (const float*)d_in[3];
  const float* Wb  = (const float*)d_in[4];
  const float* Wo  = (const float*)d_in[5];
  const float* cq  = (const float*)d_in[6];
  const float* ck  = (const float*)d_in[7];
  const float* cv  = (const float*)d_in[8];
  const float* onw = (const float*)d_in[9];
  float* out = (float*)d_out;

  const size_t NBTD = (size_t)B_ * T_ * D_;  // 16,777,216
  float* ws = (float*)d_ws;
  float* qA = ws;
  float* kA = qA + NBTD;
  float* vA = kA + NBTD;
  float* qB = vA + NBTD;
  float* kB = qB + NBTD;
  float* vB = kB + NBTD;
  float* betaW = vB + NBTD;  // B*T*H

  const int M = B_ * T_;       // 8192
  dim3 gG(D_ / 128, M / 128);  // (16, 64)

  // Projections (bf16 WMMA GEMMs)
  gemm_bf16<<<gG, 256, 0, stream>>>(x, Wq, qA, M, D_, D_);
  gemm_bf16<<<gG, 256, 0, stream>>>(x, Wk, kA, M, D_, D_);
  gemm_bf16<<<gG, 256, 0, stream>>>(x, Wv, vA, M, D_, D_);
  beta_kernel<<<M, 64, 0, stream>>>(x, Wb, betaW);

  // Shared causal conv + SiLU
  const int nConvBlk = (int)(NBTD / 256);
  conv_silu_kernel<<<nConvBlk, 256, 0, stream>>>(qA, cq, qB);
  conv_silu_kernel<<<nConvBlk, 256, 0, stream>>>(kA, ck, kB);
  conv_silu_kernel<<<nConvBlk, 256, 0, stream>>>(vA, cv, vB);

  // L2 norm per head; fold dk^-0.5 into q
  const int nvec = B_ * T_ * H_;  // 131072
  const int nNormBlk = nvec * 32 / 256;
  l2norm_kernel<<<nNormBlk, 256, 0, stream>>>(qB, 0.08838834764831845f, nvec);
  l2norm_kernel<<<nNormBlk, 256, 0, stream>>>(kB, 1.0f, nvec);

  // Delta rule: 32 blocks = B*H, o written into qA
  const size_t deltaLds =
      (size_t)DK * DK * 4       // SfT
      + (size_t)DK * DK * 2     // SbT
      + 3u * CH * DK * 2        // qb, kb, kbb
      + (size_t)DK * CH * 2     // kbT
      + (size_t)CH * DK * 2     // wbuf
      + (size_t)DK * CH * 2     // ubT
      + (size_t)CH * CH * 2     // attnb
      + (size_t)CH * CH * 4     // Mf
      + CH * 4;                 // betaS   => 221,440 B
  delta_kernel<<<B_ * H_, 256, deltaLds, stream>>>(qB, kB, vB, betaW, qA);

  // RMSNorm per head, then output projection
  rms_kernel<<<nNormBlk, 256, 0, stream>>>(qA, onw, nvec);
  gemm_bf16<<<gG, 256, 0, stream>>>(qA, Wo, out, M, D_, D_);
}
